// MultiHeadSelfAttention_64433099374763
// MI455X (gfx1250) — matrix operand order repaired and checked by reading, not compiled, not run on hardware
//
#include <hip/hip_runtime.h>
#include <hip/hip_bf16.h>

// ---------------------------------------------------------------------------
// MultiHeadSelfAttention for MI455X (gfx1250, wave32, WMMA, TDM)
// B=2, S=2048, D=1024, H=16, Dh=64. Causal mask, softmax.
// Pipeline: f32->bf16 convert -> QKV GEMM (bf16 WMMA, TDM double-buffered) ->
//           fused flash attention (bf16 WMMA, TDM-staged Q, reg-pipelined K/V)
// ---------------------------------------------------------------------------

#define B_SZ   2
#define S_LEN  2048
#define D_MOD  1024
#define H_CNT  16
#define DH     64
#define N3D    3072          // 3*D
#define M_ROWS 4096          // B*S

typedef __attribute__((ext_vector_type(16))) __bf16 v16bf;
typedef __attribute__((ext_vector_type(8)))  float  v8f;
typedef __attribute__((ext_vector_type(4)))  float  f32x4;
typedef __attribute__((ext_vector_type(4)))  unsigned int u32x4;
typedef __attribute__((ext_vector_type(8)))  int    i32x8;
typedef __attribute__((ext_vector_type(4)))  int    i32x4;

// Hardware f32 -> bf16 (RNE) via native conversion
__device__ __forceinline__ unsigned short f2bf(float f) {
  union { __bf16 h; unsigned short u; } c;
  c.h = (__bf16)f;
  return c.u;
}

union FragU { v16bf v; u32x4 q[2]; };

// A-matrix 16x32 bf16 fragment from row-major LDS.
// Lane L: M = L%16, half = L/16.
// VGPR0-3 : K = half*8 .. +7       VGPR4-7 : K = 16+half*8 .. +7
__device__ __forceinline__ v16bf load_frag_a(const unsigned short* base, int stride,
                                             int row, int lane) {
  int m = lane & 15, half = lane >> 4;
  FragU f;
  const unsigned short* p = base + (size_t)(row + m) * stride;
  f.q[0] = *(const u32x4*)(p + half * 8);
  f.q[1] = *(const u32x4*)(p + 16 + half * 8);
  return f.v;
}

// B-matrix 32x16 bf16 fragment from LDS stored TRANSPOSED as [n][k].
// Lane L: N = col + L%16, VGPR0-3 : K = half*8 .. +7, VGPR4-7 : K = 16+half*8 .. +7.
__device__ __forceinline__ v16bf load_frag_b(const unsigned short* base, int stride,
                                             int col, int lane) {
  int n = lane & 15, half = lane >> 4;
  FragU f;
  const unsigned short* p = base + (size_t)(col + n) * stride;
  f.q[0] = *(const u32x4*)(p + half * 8);
  f.q[1] = *(const u32x4*)(p + 16 + half * 8);
  return f.v;
}

__device__ __forceinline__ v8f wmma_bf16(v16bf a, v16bf b, v8f c) {
  return __builtin_amdgcn_wmma_f32_16x16x32_bf16(false, a, false, b, (short)0, c,
                                                 false, false);
}

// ---------------------------------------------------------------------------
// Tensor Data Mover: 2D tile Global -> LDS with in-flight LDS row padding.
// D# per CDNA5 ISA. data_size=2B. One wave issues;
// descriptors are wave-uniform -> SGPRs. pad codes: interval 0=2DW,1=4,...;
// amount = code+1 DWORDs.
// ---------------------------------------------------------------------------
#if __has_builtin(__builtin_amdgcn_tensor_load_to_lds)
#define HAVE_TDM 1
#else
#define HAVE_TDM 0
#endif

__device__ __forceinline__ void tdm_load_2d_bf16(
    const void* gptr, const void* ldsptr,
    unsigned tile_d0, unsigned tile_d1,
    unsigned tensor_d0, unsigned tensor_d1,
    unsigned stride0_elems,
    unsigned pad_interval_code, unsigned pad_amount_code) {
#if HAVE_TDM
  unsigned long long ga = (unsigned long long)gptr;
  unsigned lds = (unsigned)(unsigned long long)ldsptr;   // low 32 = LDS offset
  u32x4 g0;
  g0.x = 1u;                                             // count=1, user mode
  g0.y = lds;                                            // lds_addr
  g0.z = (unsigned)ga;                                   // global_addr[31:0]
  g0.w = (unsigned)((ga >> 32) & 0x1FFFFFFull) | (2u << 30);  // ga[56:32], type=2
  i32x8 g1;
  g1[0] = (int)((1u << 16)                               // data_size = 2B
              | (1u << 20)                               // pad_enable
              | (pad_interval_code << 22)
              | (pad_amount_code << 25));
  g1[1] = (int)((tensor_d0 & 0xFFFFu) << 16);            // dim0[15:0]
  g1[2] = (int)(((tensor_d0 >> 16) & 0xFFFFu) | ((tensor_d1 & 0xFFFFu) << 16));
  g1[3] = (int)(((tensor_d1 >> 16) & 0xFFFFu) | ((tile_d0 & 0xFFFFu) << 16));
  g1[4] = (int)(tile_d1 & 0xFFFFu);                      // tile_dim1, tile_dim2=0
  g1[5] = (int)stride0_elems;                            // dim0_stride[31:0]
  g1[6] = 0;
  g1[7] = 0;
  i32x4 z4 = {0, 0, 0, 0};
#if __clang_major__ >= 23
  i32x8 z8 = {0, 0, 0, 0, 0, 0, 0, 0};
  __builtin_amdgcn_tensor_load_to_lds(g0, g1, z4, z4, z8, 0);
#else
  __builtin_amdgcn_tensor_load_to_lds(g0, g1, z4, z4, 0);
#endif
#endif
}

__device__ __forceinline__ void tdm_wait() {
#if __has_builtin(__builtin_amdgcn_s_wait_tensorcnt)
  __builtin_amdgcn_s_wait_tensorcnt(0);
#endif
}

// ---------------------------------------------------------------------------
// Kernel 1: fp32 -> bf16 convert, 8 elements/thread, B128 in / B128 out
// ---------------------------------------------------------------------------
__global__ void f32_to_bf16_kernel(const float* __restrict__ in,
                                   unsigned short* __restrict__ out, int n) {
  int i = (blockIdx.x * blockDim.x + threadIdx.x) * 8;
  int stride = gridDim.x * blockDim.x * 8;
  for (; i < n; i += stride) {
    f32x4 a = *(const f32x4*)(in + i);
    f32x4 b = *(const f32x4*)(in + i + 4);
    union { unsigned short us[8]; u32x4 v; } o;
#pragma unroll
    for (int e = 0; e < 4; ++e) o.us[e] = f2bf(a[e]);
#pragma unroll
    for (int e = 0; e < 4; ++e) o.us[4 + e] = f2bf(b[e]);
    *(u32x4*)(out + i) = o.v;
  }
}

// ---------------------------------------------------------------------------
// Kernel 2: QKV GEMM  C[4096,3072] = A[4096,1024] * B[1024,3072]
// bf16 in/out, fp32 accumulate. Block 128x128, BK=32, 8 waves x (4x2) tiles.
// Double-buffered LDS: TDM for A tile kt+1 and global B loads for kt+1 are in
// flight while the 8 WMMAs of tile kt execute; s_wait_tensorcnt after compute.
// ---------------------------------------------------------------------------
#define GM 128
#define GN 128
#define GK 32
#define ASTR 48
#define NKT (D_MOD / GK)

__global__ __launch_bounds__(256) void qkv_gemm_kernel(
    const unsigned short* __restrict__ A, const unsigned short* __restrict__ Bm,
    unsigned short* __restrict__ C) {
  __shared__ __align__(16) unsigned short As[2 * GM * ASTR];
  __shared__ __align__(16) unsigned short Bs[2 * GN * ASTR];  // transposed [n][k]

  int tid = threadIdx.x, lane = tid & 31, w = tid >> 5;
  int m0 = blockIdx.y * GM, n0 = blockIdx.x * GN;
  int wm = (w >> 2) * 64;
  int wn = (w & 3) * 32;

  v8f acc[4][2] = {};
  u32x4 breg[2];

  auto stage_a = [&](unsigned short* Adst, int kt2) {
#if HAVE_TDM
    if (w == 0)
      tdm_load_2d_bf16(&A[(size_t)m0 * D_MOD + kt2 * GK], Adst,
                       GK, GM, GK, GM, D_MOD, /*pad*/ 3, 7);
#else
#pragma unroll
    for (int i = 0; i < 2; ++i) {
      int c = tid + 256 * i;
      int r = c >> 2, cc = (c & 3) * 8;
      *(u32x4*)(&Adst[r * ASTR + cc]) =
          *(const u32x4*)(&A[(size_t)(m0 + r) * D_MOD + kt2 * GK + cc]);
    }
#endif
  };
  auto stage_a_wait = [&]() {
#if HAVE_TDM
    if (w == 0) tdm_wait();
#endif
  };
  auto load_b_regs = [&](int kt2) {
#pragma unroll
    for (int i = 0; i < 2; ++i) {
      int c = tid + 256 * i;
      int k = c >> 4, cn = (c & 15) * 8;
      breg[i] = *(const u32x4*)(&Bm[(size_t)(kt2 * GK + k) * N3D + n0 + cn]);
    }
  };
  auto store_b_lds = [&](unsigned short* Bdst) {
#pragma unroll
    for (int i = 0; i < 2; ++i) {
      int c = tid + 256 * i;
      int k = c >> 4, cn = (c & 15) * 8;
      union { u32x4 q; unsigned short u[8]; } t2;
      t2.q = breg[i];
#pragma unroll
      for (int e = 0; e < 8; ++e) Bdst[(cn + e) * ASTR + k] = t2.u[e];
    }
  };

  // Prologue: stage tile 0
  stage_a(As, 0);
  load_b_regs(0);
  store_b_lds(Bs);
  stage_a_wait();
  __syncthreads();

  for (int kt = 0; kt < NKT; ++kt) {
    unsigned short* Acur = As + (kt & 1) * (GM * ASTR);
    unsigned short* Bcur = Bs + (kt & 1) * (GN * ASTR);
    unsigned short* Anxt = As + ((kt + 1) & 1) * (GM * ASTR);
    unsigned short* Bnxt = Bs + ((kt + 1) & 1) * (GN * ASTR);
    bool has_next = (kt + 1) < NKT;

    // Put next tile's DMA + global loads in flight before compute
    if (has_next) {
      stage_a(Anxt, kt + 1);
      load_b_regs(kt + 1);
    }

    // Compute on current tile (8 WMMAs) while next tile streams in
    v16bf af[4], bf[2];
#pragma unroll
    for (int i = 0; i < 4; ++i) af[i] = load_frag_a(Acur, ASTR, wm + i * 16, lane);
#pragma unroll
    for (int j = 0; j < 2; ++j) bf[j] = load_frag_b(Bcur, ASTR, wn + j * 16, lane);
#pragma unroll
    for (int i = 0; i < 4; ++i)
#pragma unroll
      for (int j = 0; j < 2; ++j) acc[i][j] = wmma_bf16(af[i], bf[j], acc[i][j]);

    if (has_next) {
      store_b_lds(Bnxt);
      stage_a_wait();
    }
    __syncthreads();
  }

  int cl = lane & 15, half = lane >> 4;
#pragma unroll
  for (int i = 0; i < 4; ++i)
#pragma unroll
    for (int j = 0; j < 2; ++j)
#pragma unroll
      for (int r = 0; r < 8; ++r) {
        int row = m0 + wm + i * 16 + r + 8 * half;
        int col = n0 + wn + j * 16 + cl;
        C[(size_t)row * N3D + col] = f2bf(acc[i][j][r]);
      }
}

// ---------------------------------------------------------------------------
// Kernel 3: fused causal flash attention. Grid (S/128, B*H), 8 waves,
// wave = 16 query rows, key blocks of 64. Scrambled K view:
//   k[i, jd] = qkv[b, (i%32)*64 + jd, 1024 + h*64 + i/32]
// Q tile staged by TDM. K gather + V rows for block kb+1 are register-
// pipelined under block kb's WMMA/softmax work.
// ---------------------------------------------------------------------------
#define BQ 128
#define BK 64
#define QSTR 72
#define KSTR 72
#define VSTR 72
#define PSTR 72
#define NEG_BIG (-1.0e9f)

__global__ __launch_bounds__(256) void attn_kernel(
    const unsigned short* __restrict__ qkv, float* __restrict__ out) {
  __shared__ __align__(16) unsigned short Qs[BQ * QSTR];        // [qrow][d]
  __shared__ __align__(16) unsigned short Ks[BK * KSTR];        // [key][d]
  __shared__ __align__(16) unsigned short VTs[DH * VSTR];       // [d][key]
  __shared__ __align__(16) unsigned short Ps[8 * 16 * PSTR];    // per-wave P

  int qb = blockIdx.x;
  int bh = blockIdx.y;
  int b = bh >> 4, h = bh & 15;
  int tid = threadIdx.x, lane = tid & 31, w = tid >> 5;
  int half = lane >> 4;
  int qrow_base = qb * BQ + w * 16;

  // Stage Q tile (128 x 64)
#if HAVE_TDM
  if (w == 0)
    tdm_load_2d_bf16(&qkv[((size_t)(b * S_LEN + qb * BQ)) * N3D + h * DH], Qs,
                     DH, BQ, DH, BQ, N3D, /*pad*/ 4, 3);
#else
#pragma unroll
  for (int i = 0; i < 4; ++i) {
    int c = tid + 256 * i;
    int r = c >> 3, cc = (c & 7) * 8;
    *(u32x4*)(&Qs[r * QSTR + cc]) =
        *(const u32x4*)(&qkv[((size_t)(b * S_LEN + qb * BQ + r)) * N3D + h * DH + cc]);
  }
#endif

  float mrow[8], lrow[8];
  v8f acco[4] = {};
#pragma unroll
  for (int r = 0; r < 8; ++r) { mrow[r] = -3.0e38f; lrow[r] = 0.0f; }

  // Staging geometry (fixed per thread)
  int kr = tid >> 2;                 // key row 0..63 (K gather)
  int kc0 = (tid & 3) * 16;
  int vr = tid >> 3;                 // for V: rows via chunks below
  (void)vr;

  unsigned short kreg[16];
  u32x4 vreg[2];

  auto load_k_regs = [&](int kb2) {
    int kg = kb2 * BK + kr;
    int dprime = kg >> 5;
    int sbase = (kg & 31) * 64;
#pragma unroll
    for (int j = 0; j < 16; ++j)
      kreg[j] = qkv[((size_t)(b * S_LEN + sbase + kc0 + j)) * N3D + D_MOD +
                    h * DH + dprime];
  };
  auto load_v_regs = [&](int kb2) {
#pragma unroll
    for (int i = 0; i < 2; ++i) {
      int c = tid + 256 * i;
      int r = c >> 3, cc = (c & 7) * 8;
      vreg[i] = *(const u32x4*)(
          &qkv[((size_t)(b * S_LEN + kb2 * BK + r)) * N3D + 2 * D_MOD + h * DH + cc]);
    }
  };

  int kb_max = 2 * qb + 1;
  load_k_regs(0);
  load_v_regs(0);

  for (int kb = 0; kb <= kb_max; ++kb) {
    __syncthreads();
    // Commit pipelined registers -> LDS tiles
#pragma unroll
    for (int j = 0; j < 16; ++j) Ks[kr * KSTR + kc0 + j] = kreg[j];
#pragma unroll
    for (int i = 0; i < 2; ++i) {
      int c = tid + 256 * i;
      int r = c >> 3, cc = (c & 7) * 8;
      union { u32x4 q; unsigned short u[8]; } tv;
      tv.q = vreg[i];
#pragma unroll
      for (int e = 0; e < 8; ++e) VTs[(cc + e) * VSTR + r] = tv.u[e];
    }
    // Issue next block's global loads; they retire under this block's compute
    if (kb < kb_max) {
      load_k_regs(kb + 1);
      load_v_regs(kb + 1);
    }
    if (kb + 2 <= kb_max) {
      int kg2 = (kb + 2) * BK + kr;
      __builtin_prefetch(
          &qkv[((size_t)(b * S_LEN + (kg2 & 31) * 64 + kc0)) * N3D + D_MOD +
               h * DH + (kg2 >> 5)], 0, 0);
    }
#if HAVE_TDM
    if (kb == 0 && w == 0) tdm_wait();   // Q tile DMA done before first use
#endif
    __syncthreads();

    // --- S = Q * K^T  (wave: 16 q-rows x 64 keys, fp32 acc) ---
    v8f sc[4] = {};
#pragma unroll
    for (int ks = 0; ks < 2; ++ks) {
      v16bf a = load_frag_a(Qs + ks * 32, QSTR, w * 16, lane);
#pragma unroll
      for (int t = 0; t < 4; ++t) {
        v16bf bm = load_frag_b(Ks + ks * 32, KSTR, t * 16, lane);
        sc[t] = wmma_bf16(a, bm, sc[t]);
      }
    }

    // --- scale + causal mask + online softmax ---
    float mnew[8];
#pragma unroll
    for (int r = 0; r < 8; ++r) {
      int qrow = qrow_base + r + 8 * half;
      float mx = -3.0e38f;
#pragma unroll
      for (int t = 0; t < 4; ++t) {
        int key = kb * BK + t * 16 + (lane & 15);
        float s = sc[t][r] * 0.125f;
        if (key > qrow) s = NEG_BIG;
        sc[t][r] = s;
        mx = fmaxf(mx, s);
      }
#pragma unroll
      for (int d = 1; d < 16; d <<= 1) mx = fmaxf(mx, __shfl_xor(mx, d, 32));
      mnew[r] = fmaxf(mrow[r], mx);
    }

    // --- P = exp(S - m), row sums, stage P (A-layout via LDS) ---
    unsigned short* pw = Ps + w * 16 * PSTR;
#pragma unroll
    for (int r = 0; r < 8; ++r) {
      int prow = r + 8 * half;
      float sum = 0.0f;
#pragma unroll
      for (int t = 0; t < 4; ++t) {
        float p = __expf(sc[t][r] - mnew[r]);
        sum += p;
        pw[prow * PSTR + t * 16 + (lane & 15)] = f2bf(p);
      }
#pragma unroll
      for (int d = 1; d < 16; d <<= 1) sum += __shfl_xor(sum, d, 32);
      float corr = __expf(mrow[r] - mnew[r]);
      lrow[r] = lrow[r] * corr + sum;
      mrow[r] = mnew[r];
#pragma unroll
      for (int t = 0; t < 4; ++t) acco[t][r] *= corr;
    }

    // --- O += P (16x64) * V (64x64) ---
#pragma unroll
    for (int ks = 0; ks < 2; ++ks) {
      v16bf a = load_frag_a(pw + ks * 32, PSTR, 0, lane);
#pragma unroll
      for (int t = 0; t < 4; ++t) {
        v16bf bm = load_frag_b(VTs + ks * 32, VSTR, t * 16, lane);
        acco[t] = wmma_bf16(a, bm, acco[t]);
      }
    }
  }

  // --- normalize and write out[b, qrow, h*64 + d] (fp32) ---
#pragma unroll
  for (int r = 0; r < 8; ++r) {
    int qrow = qrow_base + r + 8 * half;
    float inv = 1.0f / lrow[r];
#pragma unroll
    for (int t = 0; t < 4; ++t) {
      int d = t * 16 + (lane & 15);
      out[((size_t)(b * S_LEN + qrow)) * D_MOD + h * DH + d] = acco[t][r] * inv;
    }
  }
}

// ---------------------------------------------------------------------------
extern "C" void kernel_launch(void* const* d_in, const int* in_sizes, int n_in,
                              void* d_out, int out_size, void* d_ws, size_t ws_size,
                              hipStream_t stream) {
  const float* x  = (const float*)d_in[0];   // (2, 2048, 1024)
  const float* wq = (const float*)d_in[1];   // (1024, 3072)
  float* out = (float*)d_out;                // (2, 2048, 1024)

  unsigned short* xb   = (unsigned short*)d_ws;                 // 4096x1024 bf16
  unsigned short* wb   = xb + (size_t)M_ROWS * D_MOD;           // 1024x3072 bf16
  unsigned short* qkvb = wb + (size_t)D_MOD * N3D;              // 4096x3072 bf16

  f32_to_bf16_kernel<<<1024, 256, 0, stream>>>(x,  xb, M_ROWS * D_MOD);
  f32_to_bf16_kernel<<<1024, 256, 0, stream>>>(wq, wb, D_MOD * N3D);

  qkv_gemm_kernel<<<dim3(N3D / GN, M_ROWS / GM), 256, 0, stream>>>(xb, wb, qkvb);

  attn_kernel<<<dim3(S_LEN / BQ, B_SZ * H_CNT), 256, 0, stream>>>(qkvb, out);
}
